// RNN_16011638080207
// MI455X (gfx1250) — matrix operand order repaired and checked by reading, not compiled, not run on hardware
//
#include <hip/hip_runtime.h>

// ---------------------------------------------------------------------------
// Elman RNN on gfx1250 via v_wmma_f32_16x16x32_f16 (f16 operands, f32 accum).
//   hidden[t] = relu( x[t]@Wi^T + bi + hidden[t-1]@Wh^T + bh )
//   y[t]      = hidden[t]@Wo^T + bo
// Outputs concatenated: hidden [T,B,H] | hT [1,B,H] | y [T,B,O]
//
// Optimization vs round 1:
//  * Weights pre-converted f32->f16 once into d_ws (removes 32/40 cvts per
//    K-step); scan keeps an f16 double-buffered shadow of h_t so the serial
//    recurrence issues only b128 loads + WMMA in the K-loop.
//  * Scan GEMM uses 4-way K-split across 4 waves/workgroup with LDS
//    reduction (512 waves/step instead of 128 on the latency-critical chain).
// ---------------------------------------------------------------------------

#define RNN_T    256
#define RNN_B    128
#define RNN_DIN  512
#define RNN_DH   1024
#define RNN_DOUT 512

typedef __attribute__((ext_vector_type(16))) _Float16 v16h;
typedef __attribute__((ext_vector_type(8)))  _Float16 v8h;
typedef __attribute__((ext_vector_type(8)))  float    v8f;
typedef __attribute__((ext_vector_type(4)))  float    v4f;

// ---------------- fragment loaders -----------------------------------------
// A (16x32, MxK): lane holds row m=lane%16; kb=16*(lane/16);
//   halves 0..15 <- K=kb..kb+15.
__device__ __forceinline__ v16h load_frag_a32(const float* __restrict__ p) {
  v4f u0 = *(const v4f*)(p + 0);
  v4f u1 = *(const v4f*)(p + 4);
  v4f u2 = *(const v4f*)(p + 8);
  v4f u3 = *(const v4f*)(p + 12);
  v16h r;
#pragma unroll
  for (int j = 0; j < 4; ++j) {
    r[j] = (_Float16)u0[j]; r[4+j] = (_Float16)u1[j];
    r[8+j] = (_Float16)u2[j]; r[12+j] = (_Float16)u3[j];
  }
  return r;
}
__device__ __forceinline__ v16h load_frag_a16(const _Float16* __restrict__ p) {
  v8h lo = *(const v8h*)(p + 0);
  v8h hi = *(const v8h*)(p + 8);
  v16h r;
#pragma unroll
  for (int j = 0; j < 8; ++j) { r[j] = lo[j]; r[8+j] = hi[j]; }
  return r;
}
// B (32x16, KxN): lane holds col n=lane%16; halves 0..15 <- K=16*(lane/16)+0..15.
__device__ __forceinline__ v16h load_frag_b32(const float* __restrict__ p) {
  v4f u0 = *(const v4f*)(p + 0);
  v4f u1 = *(const v4f*)(p + 4);
  v4f u2 = *(const v4f*)(p + 8);
  v4f u3 = *(const v4f*)(p + 12);
  v16h r;
#pragma unroll
  for (int j = 0; j < 4; ++j) {
    r[j] = (_Float16)u0[j]; r[4+j] = (_Float16)u1[j];
    r[8+j] = (_Float16)u2[j]; r[12+j] = (_Float16)u3[j];
  }
  return r;
}
__device__ __forceinline__ v16h load_frag_b16(const _Float16* __restrict__ p) {
  v8h lo = *(const v8h*)(p + 0);
  v8h hi = *(const v8h*)(p + 8);
  v16h r;
#pragma unroll
  for (int j = 0; j < 8; ++j) { r[j] = lo[j]; r[8+j] = hi[j]; }
  return r;
}

// ---------------- epilogue for one 16x16 subtile ----------------------------
__device__ __forceinline__ void epilogue_tile(
    const v8f& a, int s, int m0, int n0, int l16, int halfsel,
    const float* __restrict__ bias, const float* __restrict__ Z,
    float* __restrict__ D, float* __restrict__ D2,
    _Float16* __restrict__ H16, int N, int relu)
{
  const int n = n0 + 16 * s + l16;
  const float bn = bias ? bias[n] : 0.0f;
#pragma unroll
  for (int v = 0; v < 8; ++v) {
    const int m = m0 + v + 8 * halfsel;
    const size_t idx = (size_t)m * (size_t)N + (size_t)n;
    float val = a[v] + bn;
    if (Z)    val += Z[idx];
    if (relu) val = val > 0.0f ? val : 0.0f;
    D[idx] = val;
    if (D2)  D2[idx] = val;
    if (H16) H16[idx] = (_Float16)val;
  }
}

// ---------------- GEMM: D = act(A @ Bw^T + bias [+ Z]) ---------------------
// A: [M,K] row-major (f32 or f16 per A16); nullptr -> zero matrix.
// Bw: [N,K] row-major (f32 or f16 per B16).  D/Z stride = N.
// Block = 32*KWAVES threads; wave w accumulates K-slice [w*K/KW, (w+1)*K/KW),
// partials reduced through LDS; wave w finalizes subtile s = w (KWAVES==4).
// grid = (N/64, M/16).
template<bool A16, bool B16, int KWAVES>
__global__ __launch_bounds__(32 * KWAVES)
void rnn_wmma_gemm(const void* __restrict__ Ap, int lda,
                   const void* __restrict__ Bp, int ldb,
                   const float* __restrict__ bias,
                   const float* __restrict__ Z,
                   float* __restrict__ D,
                   float* __restrict__ D2,
                   _Float16* __restrict__ H16,
                   int N, int K, int relu)
{
  static_assert(KWAVES == 1 || KWAVES == 4, "KWAVES must be 1 or 4");
  constexpr int LSTRIDE = 36;   // DWORD stride pad: <=2-way conflict on 64 banks

  const int tid     = (int)threadIdx.x;
  const int lane    = tid & 31;
  const int w       = tid >> 5;
  const int l16     = lane & 15;
  const int halfsel = lane >> 4;
  const int m0      = blockIdx.y * 16;
  const int n0      = blockIdx.x * 64;

  v8f acc[4] = {};

  if (Ap != nullptr) {
    const int kslice = K / KWAVES;
    const int kbeg   = w * kslice;
    const int kend   = kbeg + kslice;

    const size_t a_off = (size_t)(m0 + l16) * (size_t)lda + (size_t)(halfsel * 16);
    size_t b_off[4];
#pragma unroll
    for (int s = 0; s < 4; ++s)
      b_off[s] = (size_t)(n0 + 16 * s + l16) * (size_t)ldb + (size_t)(halfsel * 16);

    const float*    A32 = (const float*)Ap;
    const _Float16* Ah  = (const _Float16*)Ap;
    const float*    B32 = (const float*)Bp;
    const _Float16* Bh  = (const _Float16*)Bp;

    for (int k0 = kbeg; k0 < kend; k0 += 32) {
      v16h a;
      if constexpr (A16) a = load_frag_a16(Ah + a_off + k0);
      else               a = load_frag_a32(A32 + a_off + k0);
#pragma unroll
      for (int s = 0; s < 4; ++s) {
        v16h b;
        if constexpr (B16) b = load_frag_b16(Bh + b_off[s] + k0);
        else               b = load_frag_b32(B32 + b_off[s] + k0);
        acc[s] = __builtin_amdgcn_wmma_f32_16x16x32_f16(
                     false, a, false, b, (short)0, acc[s], false, false);
      }
    }
  }

  if constexpr (KWAVES == 4) {
    __shared__ float red[4 * 32 * LSTRIDE];
    float* myred = &red[(w * 32 + lane) * LSTRIDE];
#pragma unroll
    for (int s = 0; s < 4; ++s) {
      v4f lo, hi;
#pragma unroll
      for (int j = 0; j < 4; ++j) { lo[j] = acc[s][j]; hi[j] = acc[s][j + 4]; }
      *(v4f*)(myred + s * 8)     = lo;
      *(v4f*)(myred + s * 8 + 4) = hi;
    }
    __syncthreads();
    const int s = w;                         // each wave finalizes one subtile
    v8f sum = {};
#pragma unroll
    for (int ww = 0; ww < 4; ++ww) {
      const float* rp = &red[(ww * 32 + lane) * LSTRIDE + s * 8];
      v4f lo = *(const v4f*)(rp);
      v4f hi = *(const v4f*)(rp + 4);
#pragma unroll
      for (int j = 0; j < 4; ++j) { sum[j] += lo[j]; sum[4 + j] += hi[j]; }
    }
    epilogue_tile(sum, s, m0, n0, l16, halfsel, bias, Z, D, D2, H16, N, relu);
  } else {
#pragma unroll
    for (int s = 0; s < 4; ++s)
      epilogue_tile(acc[s], s, m0, n0, l16, halfsel, bias, Z, D, D2, H16, N, relu);
  }
}

// ---------------- f32 -> f16 weight conversion ------------------------------
__global__ __launch_bounds__(256)
void cvt_f32_to_f16(const float* __restrict__ src, _Float16* __restrict__ dst, int n8) {
  const int i = (int)(blockIdx.x * 256 + threadIdx.x);
  if (i < n8) {
    v4f a = *(const v4f*)(src + (size_t)i * 8);
    v4f b = *(const v4f*)(src + (size_t)i * 8 + 4);
    v8h o;
#pragma unroll
    for (int j = 0; j < 4; ++j) { o[j] = (_Float16)a[j]; o[4 + j] = (_Float16)b[j]; }
    *(v8h*)(dst + (size_t)i * 8) = o;
  }
}

// ---------------------------------------------------------------------------
extern "C" void kernel_launch(void* const* d_in, const int* in_sizes, int n_in,
                              void* d_out, int out_size, void* d_ws, size_t ws_size,
                              hipStream_t stream) {
  (void)in_sizes; (void)n_in; (void)out_size;

  const float* x  = (const float*)d_in[0];   // [T,B,DIN]
  const float* Wi = (const float*)d_in[1];   // [DH,DIN]
  const float* bi = (const float*)d_in[2];   // [DH]
  const float* Wh = (const float*)d_in[3];   // [DH,DH]
  const float* bh = (const float*)d_in[4];   // [DH]
  const float* Wo = (const float*)d_in[5];   // [DOUT,DH]
  const float* bo = (const float*)d_in[6];   // [DOUT]

  float* out    = (float*)d_out;
  float* hidden = out;                                     // [T,B,DH]
  float* hT     = out + (size_t)RNN_T * RNN_B * RNN_DH;    // [B,DH]
  float* y      = hT  + (size_t)RNN_B * RNN_DH;            // [T,B,DOUT]

  // workspace layout (f16 weights + double-buffered f16 hidden)
  const size_t WI_B  = (size_t)RNN_DH  * RNN_DIN * 2;      // 1 MB
  const size_t WH_B  = (size_t)RNN_DH  * RNN_DH  * 2;      // 2 MB
  const size_t WO_B  = (size_t)RNN_DOUT* RNN_DH  * 2;      // 1 MB
  const size_t H16_B = (size_t)RNN_B   * RNN_DH  * 2;      // 256 KB
  const size_t WS_NEED = WI_B + WH_B + WO_B + 2 * H16_B;

  const bool f16w = (d_ws != nullptr) && (ws_size >= WS_NEED);

  const dim3 blk1(32, 1, 1);
  const dim3 blk4(128, 1, 1);

  if (f16w) {
    char* ws = (char*)d_ws;
    _Float16* Wi16 = (_Float16*)(ws);
    _Float16* Wh16 = (_Float16*)(ws + WI_B);
    _Float16* Wo16 = (_Float16*)(ws + WI_B + WH_B);
    _Float16* h16a = (_Float16*)(ws + WI_B + WH_B + WO_B);
    _Float16* h16b = (_Float16*)(ws + WI_B + WH_B + WO_B + H16_B);
    _Float16* h16buf[2] = { h16a, h16b };

    // 0) convert weights once per call
    {
      int n8;
      n8 = RNN_DH * RNN_DIN / 8;
      cvt_f32_to_f16<<<dim3((n8 + 255) / 256), dim3(256), 0, stream>>>(Wi, Wi16, n8);
      n8 = RNN_DH * RNN_DH / 8;
      cvt_f32_to_f16<<<dim3((n8 + 255) / 256), dim3(256), 0, stream>>>(Wh, Wh16, n8);
      n8 = RNN_DOUT * RNN_DH / 8;
      cvt_f32_to_f16<<<dim3((n8 + 255) / 256), dim3(256), 0, stream>>>(Wo, Wo16, n8);
    }

    // 1) zi = x @ Wi^T + bi -> hidden (no relu)
    {
      const int M = RNN_T * RNN_B, N = RNN_DH, K = RNN_DIN;
      dim3 grid(N / 64, M / 16, 1);
      rnn_wmma_gemm<false, true, 1><<<grid, blk1, 0, stream>>>(
          x, K, Wi16, K, bi, nullptr, hidden, nullptr, nullptr, N, K, 0);
    }

    // 2) scan: hidden[t] = relu(hidden[t] + h_{t-1} @ Wh^T + bh); f16 shadow of h_t
    {
      const int M = RNN_B, N = RNN_DH, K = RNN_DH;
      dim3 grid(N / 64, M / 16, 1);
      for (int t = 0; t < RNN_T; ++t) {
        const _Float16* hprev = (t == 0) ? nullptr : h16buf[(t - 1) & 1];
        float* zt = hidden + (size_t)t * RNN_B * RNN_DH;
        float* d2 = (t == RNN_T - 1) ? hT : nullptr;
        rnn_wmma_gemm<true, true, 4><<<grid, blk4, 0, stream>>>(
            hprev, K, Wh16, K, bh, zt, zt, d2, h16buf[t & 1], N, K, 1);
      }
    }

    // 3) y = hidden @ Wo^T + bo
    {
      const int M = RNN_T * RNN_B, N = RNN_DOUT, K = RNN_DH;
      dim3 grid(N / 64, M / 16, 1);
      rnn_wmma_gemm<false, true, 1><<<grid, blk1, 0, stream>>>(
          hidden, K, Wo16, K, bo, nullptr, y, nullptr, nullptr, N, K, 0);
    }
  } else {
    // ---- fallback: no workspace; load f32 operands, convert in-flight ----
    {
      const int M = RNN_T * RNN_B, N = RNN_DH, K = RNN_DIN;
      dim3 grid(N / 64, M / 16, 1);
      rnn_wmma_gemm<false, false, 1><<<grid, blk1, 0, stream>>>(
          x, K, Wi, K, bi, nullptr, hidden, nullptr, nullptr, N, K, 0);
    }
    {
      const int M = RNN_B, N = RNN_DH, K = RNN_DH;
      dim3 grid(N / 64, M / 16, 1);
      for (int t = 0; t < RNN_T; ++t) {
        const float* hprev = (t == 0) ? nullptr
                                      : (hidden + (size_t)(t - 1) * RNN_B * RNN_DH);
        float* zt = hidden + (size_t)t * RNN_B * RNN_DH;
        float* d2 = (t == RNN_T - 1) ? hT : nullptr;
        rnn_wmma_gemm<false, false, 4><<<grid, blk4, 0, stream>>>(
            hprev, K, Wh, K, bh, zt, zt, d2, nullptr, N, K, 1);
      }
    }
    {
      const int M = RNN_T * RNN_B, N = RNN_DOUT, K = RNN_DH;
      dim3 grid(N / 64, M / 16, 1);
      rnn_wmma_gemm<false, false, 1><<<grid, blk1, 0, stream>>>(
          hidden, K, Wo, K, bo, nullptr, y, nullptr, nullptr, N, K, 0);
    }
  }
}
